// HierarchicalMemory_20229295964723
// MI455X (gfx1250) — compile-verified, matrix operand order checked by reading, not run on hardware
//
#include <hip/hip_runtime.h>
#include <cstdint>
#include <cstddef>

// Problem dims (fixed by the reference)
#define Hd 1024      // hidden
#define Ms 8192      // memory slots
#define Tt 2048      // B*S tokens
#define THRs 0.7f
#define UPDr 0.05f
#define INV_TEMP 10.0f

typedef __attribute__((ext_vector_type(16))) _Float16 v16h;
typedef __attribute__((ext_vector_type(8)))  _Float16 v8h;
typedef __attribute__((ext_vector_type(8)))  float    v8f;

// ---------------- conversions ----------------
__global__ void k_f32_to_f16(const float* __restrict__ src, _Float16* __restrict__ dst, int n) {
  int i = blockIdx.x * blockDim.x + threadIdx.x;
  if (i < n) dst[i] = (_Float16)src[i];
}

// dst[C x R] = transpose(src[R x C]), f32 -> f16, LDS-tiled for coalescing.
// Grid: (C/32, R/32), 256 threads.
__global__ __launch_bounds__(256)
void k_transpose_f16(const float* __restrict__ src, _Float16* __restrict__ dst,
                     int R, int C) {
  __shared__ float tile[32][33];
  const int c0 = blockIdx.x * 32, r0 = blockIdx.y * 32;
  const int tx = threadIdx.x & 31, ty = threadIdx.x >> 5;   // ty: 0..7
  #pragma unroll
  for (int i = 0; i < 4; ++i)
    tile[ty + 8 * i][tx] = src[(size_t)(r0 + ty + 8 * i) * C + (c0 + tx)];
  __syncthreads();
  #pragma unroll
  for (int i = 0; i < 4; ++i)
    dst[(size_t)(c0 + ty + 8 * i) * R + (r0 + tx)] = (_Float16)tile[tx][ty + 8 * i];
}

// per-row L2 normalize; optional f32 / f16 outputs, optional raw sum-of-squares out
__global__ void k_l2norm_rows(const float* __restrict__ src, float* __restrict__ dst32,
                              _Float16* __restrict__ dst16, float* __restrict__ norm2_out,
                              int cols) {
  __shared__ float sh[256];
  const int tid = threadIdx.x;
  const float* r = src + (size_t)blockIdx.x * cols;
  float ss = 0.0f;
  for (int i = tid; i < cols; i += 256) { float v = r[i]; ss += v * v; }
  sh[tid] = ss; __syncthreads();
  for (int s = 128; s; s >>= 1) { if (tid < s) sh[tid] += sh[tid + s]; __syncthreads(); }
  float n2 = sh[0];
  if (norm2_out && tid == 0) norm2_out[blockIdx.x] = n2;
  float rn = 1.0f / fmaxf(sqrtf(n2), 1e-12f);
  for (int i = tid; i < cols; i += 256) {
    float v = r[i] * rn;
    if (dst32) dst32[(size_t)blockIdx.x * cols + i] = v;
    if (dst16) dst16[(size_t)blockIdx.x * cols + i] = (_Float16)v;
  }
}

// ---------------- WMMA GEMM: C[M,N] = alpha * A[M,K] @ B[K,N] + bias ----------------
// 64x64 output per wave: 4 A-fragments x 4 B-fragments -> 16 v_wmma per K-step
// (16 loads : 16 wmma instead of 2:1), 4x L2-traffic reduction vs 16x16/wave.
// A row-major f16 [M,K]; Bt row-major [N,K] f16 (B transposed) so every fragment
// load is a contiguous v8h per the CDNA5 16-bit A-matrix layout:
//   lane l (r = l&15, h = l>>4): frag[0..7]  = K = k0 + h*8 + 0..7
//                                frag[8..15] = K = k0 + 16 + h*8 + 0..7
__global__ __launch_bounds__(32)
void k_gemm_f16(const _Float16* __restrict__ A, const _Float16* __restrict__ Bt,
                const float* __restrict__ bias, float* __restrict__ C,
                int Nc, int K, float alpha) {
  const int lane = threadIdx.x;
  const int r = lane & 15, h = lane >> 4;
  const int tm = blockIdx.y << 6, tn = blockIdx.x << 6;

  const _Float16* ap[4];
  const _Float16* bp[4];
  #pragma unroll
  for (int i = 0; i < 4; ++i) {
    ap[i] = A  + (size_t)(tm + 16 * i + r) * K + h * 8;
    bp[i] = Bt + (size_t)(tn + 16 * i + r) * K + h * 8;
  }

  v8f acc[4][4] = {};
  for (int k0 = 0; k0 < K; k0 += 32) {
    v16h a[4], b[4];
    #pragma unroll
    for (int i = 0; i < 4; ++i) {
      v8h alo = *(const v8h*)(ap[i] + k0);
      v8h ahi = *(const v8h*)(ap[i] + k0 + 16);
      a[i] = __builtin_shufflevector(alo, ahi,
               0,1,2,3,4,5,6,7,8,9,10,11,12,13,14,15);
      v8h blo = *(const v8h*)(bp[i] + k0);
      v8h bhi = *(const v8h*)(bp[i] + k0 + 16);
      b[i] = __builtin_shufflevector(blo, bhi,
               0,1,2,3,4,5,6,7,8,9,10,11,12,13,14,15);
    }
    #pragma unroll
    for (int i = 0; i < 4; ++i)
      #pragma unroll
      for (int j = 0; j < 4; ++j)
        acc[i][j] = __builtin_amdgcn_wmma_f32_16x16x32_f16(
            false, a[i], false, b[j], (short)0, acc[i][j], false, false);
  }

  // C/D layout: VGPR e -> row (tile_m + e + 8*h), col (tile_n + (lane&15))
  float bval[4];
  #pragma unroll
  for (int j = 0; j < 4; ++j) bval[j] = bias ? bias[tn + 16 * j + r] : 0.0f;
  #pragma unroll
  for (int i = 0; i < 4; ++i) {
    #pragma unroll
    for (int j = 0; j < 4; ++j) {
      const int col = tn + 16 * j + r;
      #pragma unroll
      for (int e = 0; e < 8; ++e) {
        const int row = tm + 16 * i + e + 8 * h;
        C[(size_t)row * Nc + col] = alpha * acc[i][j][e] + bval[j];
      }
    }
  }
}

// ---------------- masked row softmax, writes f16 attn ----------------
__global__ void k_softmax_row(const float* __restrict__ sims, const float* __restrict__ usage,
                              _Float16* __restrict__ attn, int Mc) {
  __shared__ float sh[256];
  const int tid = threadIdx.x;
  const float* srow = sims + (size_t)blockIdx.x * Mc;
  _Float16* arow = attn + (size_t)blockIdx.x * Mc;
  float mx = -3.0e38f;
  for (int i = tid; i < Mc; i += 256) {
    float v = (usage[i] > 0.0f) ? srow[i] : -3.0e38f;
    mx = fmaxf(mx, v);
  }
  sh[tid] = mx; __syncthreads();
  for (int s = 128; s; s >>= 1) { if (tid < s) sh[tid] = fmaxf(sh[tid], sh[tid + s]); __syncthreads(); }
  mx = sh[0]; __syncthreads();
  float sum = 0.0f;
  for (int i = tid; i < Mc; i += 256)
    if (usage[i] > 0.0f) sum += expf(srow[i] - mx);
  sh[tid] = sum; __syncthreads();
  for (int s = 128; s; s >>= 1) { if (tid < s) sh[tid] += sh[tid + s]; __syncthreads(); }
  float inv = 1.0f / sh[0];
  for (int i = tid; i < Mc; i += 256) {
    float v = (usage[i] > 0.0f) ? expf(srow[i] - mx) * inv : 0.0f;
    arow[i] = (_Float16)v;
  }
}

// ---------------- count active slots ----------------
__global__ void k_count_active(const float* __restrict__ usage, int* __restrict__ out) {
  __shared__ int sh[256];
  int tid = threadIdx.x, c = 0;
  for (int i = tid; i < Ms; i += 256) c += (usage[i] > 0.0f) ? 1 : 0;
  sh[tid] = c; __syncthreads();
  for (int s = 128; s; s >>= 1) { if (tid < s) sh[tid] += sh[tid + s]; __syncthreads(); }
  if (tid == 0) *out = sh[0];
}

// ---------------- sequential per-token memory update ----------------
// Single persistent workgroup (32 wave32 waves). active < Ms always holds
// (4096 initial + <=2048 appends), so idx = upd ? argmax : active, and only
// rows < active can exceed THR (zero rows have sim 0). The 32MB key bank is
// L2-resident (192MB L2); global_prefetch hides row-fetch latency.
__global__ __launch_bounds__(1024)
void k_seq_update(const float* __restrict__ q,     // [Tt,Hd] normalized token keys
                  const float* __restrict__ vals,  // [Tt,Hd]
                  float* __restrict__ keys,        // [Ms,Hd] evolving
                  float* __restrict__ values,      // [Ms,Hd] evolving
                  float* __restrict__ usage,       // [Ms]
                  float* __restrict__ norm2,       // [Ms] row |key|^2, maintained
                  const int* __restrict__ active_in) {
  __shared__ float s_k[Hd];
  __shared__ float s_v[Hd];
  __shared__ float s_red[32];
  __shared__ int   s_idxred[32];
  __shared__ int   s_active, s_idx, s_upd;

  const int tid = threadIdx.x;
  const int lane = tid & 31;
  const int wave = tid >> 5;          // 0..31
  if (tid == 0) s_active = *active_in;
  __syncthreads();

  for (int t = 0; t < Tt; ++t) {
    s_k[tid] = q[(size_t)t * Hd + tid];
    s_v[tid] = vals[(size_t)t * Hd + tid];
    if (t + 1 < Tt) {
      __builtin_prefetch(q    + (size_t)(t + 1) * Hd + tid, 0, 1);
      __builtin_prefetch(vals + (size_t)(t + 1) * Hd + tid, 0, 1);
    }
    __syncthreads();
    const int active = s_active;

    // wave-per-row cosine similarity, running argmax
    float best = -3.0e38f; int bidx = -1;
    for (int i = wave; i < active; i += 32) {
      const float* kr = keys + (size_t)i * Hd;
      __builtin_prefetch(kr + 32 * Hd + lane * 32, 0, 1);  // next row for this wave
      float d = 0.0f;
      for (int e = lane; e < Hd; e += 32) d += kr[e] * s_k[e];
      #pragma unroll
      for (int off = 16; off; off >>= 1) d += __shfl_xor(d, off, 32);
      float sim = d / fmaxf(sqrtf(norm2[i]), 1e-12f);
      if (sim > best || (sim == best && i < bidx)) { best = sim; bidx = i; }
    }
    if (lane == 0) { s_red[wave] = best; s_idxred[wave] = bidx; }
    __syncthreads();

    if (tid == 0) {
      float mb = -3.0e38f; int mi = -1;
      for (int w = 0; w < 32; ++w) {
        float v = s_red[w]; int ix = s_idxred[w];
        if (ix >= 0 && (v > mb || (v == mb && ix < mi))) { mb = v; mi = ix; }
      }
      int upd = (mi >= 0) && (mb > THRs) && (usage[mi] > 0.0f);
      int idx = upd ? mi : active;
      s_upd = upd; s_idx = idx;
      if (upd) { usage[idx] = usage[idx] + 1.0f; }
      else     { usage[idx] = 1.0f; s_active = active + 1; }
    }
    __syncthreads();

    const int idx = s_idx, upd = s_upd;
    float* kr = keys   + (size_t)idx * Hd;
    float* vr = values + (size_t)idx * Hd;
    float nk, nv;
    if (upd) {
      nk = (1.0f - UPDr) * kr[tid] + UPDr * s_k[tid];
      nv = (1.0f - UPDr) * vr[tid] + UPDr * s_v[tid];
    } else {
      nk = s_k[tid]; nv = s_v[tid];
    }
    kr[tid] = nk; vr[tid] = nv;

    // new |key|^2 for the touched row
    float sq = nk * nk;
    #pragma unroll
    for (int off = 16; off; off >>= 1) sq += __shfl_xor(sq, off, 32);
    if (lane == 0) s_red[wave] = sq;
    __syncthreads();
    if (tid == 0) {
      float tot = 0.0f;
      for (int w = 0; w < 32; ++w) tot += s_red[w];
      norm2[idx] = tot;
    }
    __syncthreads();
  }
}

// ---------------- host launcher ----------------
extern "C" void kernel_launch(void* const* d_in, const int* in_sizes, int n_in,
                              void* d_out, int out_size, void* d_ws, size_t ws_size,
                              hipStream_t stream) {
  (void)in_sizes; (void)n_in; (void)out_size; (void)ws_size;
  const float* hs  = (const float*)d_in[0];
  const float* Wk  = (const float*)d_in[1];
  const float* bk  = (const float*)d_in[2];
  const float* Wv  = (const float*)d_in[3];
  const float* bv  = (const float*)d_in[4];
  const float* Wo  = (const float*)d_in[5];
  const float* bo  = (const float*)d_in[6];
  const float* mk0 = (const float*)d_in[7];
  const float* mv0 = (const float*)d_in[8];
  const float* mu0 = (const float*)d_in[9];

  float* out    = (float*)d_out;
  float* okeys  = out   + (size_t)Tt * Hd;
  float* ovals  = okeys + (size_t)Ms * Hd;
  float* ousage = ovals + (size_t)Ms * Hd;

  char* w = (char*)d_ws;
  size_t off = 0;
  auto alloc = [&](size_t bytes) -> void* {
    void* p = w + off; off += (bytes + 255) & ~(size_t)255; return p;
  };
  _Float16* hs16   = (_Float16*)alloc((size_t)Tt * Hd * 2);
  _Float16* WkT    = (_Float16*)alloc((size_t)Hd * Hd * 2);
  _Float16* WvT    = (_Float16*)alloc((size_t)Hd * Hd * 2);
  _Float16* WoT    = (_Float16*)alloc((size_t)Hd * Hd * 2);
  float*    qraw   = (float*)   alloc((size_t)Tt * Hd * 4);
  float*    q32    = (float*)   alloc((size_t)Tt * Hd * 4);
  _Float16* q16    = (_Float16*)alloc((size_t)Tt * Hd * 2);
  float*    vals   = (float*)   alloc((size_t)Tt * Hd * 4);
  _Float16* mk16   = (_Float16*)alloc((size_t)Ms * Hd * 2);
  _Float16* mvT16  = (_Float16*)alloc((size_t)Hd * Ms * 2);
  float*    sims   = (float*)   alloc((size_t)Tt * Ms * 4);
  _Float16* attn16 = (_Float16*)alloc((size_t)Tt * Ms * 2);
  float*    retr   = (float*)   alloc((size_t)Tt * Hd * 4);
  _Float16* retr16 = (_Float16*)alloc((size_t)Tt * Hd * 2);
  float*    norm2  = (float*)   alloc((size_t)Ms * 4);
  int*      activp = (int*)     alloc(256);

  const int nTH = Tt * Hd;
  k_f32_to_f16<<<(nTH + 255) / 256, 256, 0, stream>>>(hs, hs16, nTH);
  dim3 gtW(Hd / 32, Hd / 32);
  k_transpose_f16<<<gtW, 256, 0, stream>>>(Wk, WkT, Hd, Hd);
  k_transpose_f16<<<gtW, 256, 0, stream>>>(Wv, WvT, Hd, Hd);
  k_transpose_f16<<<gtW, 256, 0, stream>>>(Wo, WoT, Hd, Hd);

  dim3 gHH(Hd / 64, Tt / 64);
  k_gemm_f16<<<gHH, 32, 0, stream>>>(hs16, WkT, bk, qraw, Hd, Hd, 1.0f);
  k_gemm_f16<<<gHH, 32, 0, stream>>>(hs16, WvT, bv, vals, Hd, Hd, 1.0f);

  k_l2norm_rows<<<Tt, 256, 0, stream>>>(qraw, q32, q16, nullptr, Hd);
  k_l2norm_rows<<<Ms, 256, 0, stream>>>(mk0, nullptr, mk16, norm2, Hd);
  dim3 gtMV(Hd / 32, Ms / 32);
  k_transpose_f16<<<gtMV, 256, 0, stream>>>(mv0, mvT16, Ms, Hd);

  dim3 gSM(Ms / 64, Tt / 64);
  k_gemm_f16<<<gSM, 32, 0, stream>>>(q16, mk16, nullptr, sims, Ms, Hd, INV_TEMP);
  k_softmax_row<<<Tt, 256, 0, stream>>>(sims, mu0, attn16, Ms);
  k_gemm_f16<<<gHH, 32, 0, stream>>>(attn16, mvT16, nullptr, retr, Hd, Ms, 1.0f);
  k_f32_to_f16<<<(nTH + 255) / 256, 256, 0, stream>>>(retr, retr16, nTH);
  k_gemm_f16<<<gHH, 32, 0, stream>>>(retr16, WoT, bo, out, Hd, Hd, 1.0f);

  hipMemcpyAsync(okeys, mk0, (size_t)Ms * Hd * 4, hipMemcpyDeviceToDevice, stream);
  hipMemcpyAsync(ovals, mv0, (size_t)Ms * Hd * 4, hipMemcpyDeviceToDevice, stream);
  hipMemcpyAsync(ousage, mu0, (size_t)Ms * 4, hipMemcpyDeviceToDevice, stream);
  k_count_active<<<1, 256, 0, stream>>>(mu0, activp);
  k_seq_update<<<1, 1024, 0, stream>>>(q32, vals, okeys, ovals, ousage, norm2, activp);
}
